// Forecaster_10771777978691
// MI455X (gfx1250) — matrix operand order repaired and checked by reading, not compiled, not run on hardware
//
#include <hip/hip_runtime.h>
#include <hip/hip_bf16.h>

typedef __attribute__((ext_vector_type(16))) _Float16 v16h;
typedef __attribute__((ext_vector_type(8)))  _Float16 v8h;
typedef __attribute__((ext_vector_type(8)))  float    v8f;

// ---------------------------------------------------------------------------
// Fast activations: single v_exp_f32 + v_rcp_f32, no IEEE division expansion.
// ---------------------------------------------------------------------------
__device__ __forceinline__ float fast_rcp(float x) {
    return __builtin_amdgcn_rcpf(x);
}
__device__ __forceinline__ float sigmoid_f(float x) {
    // exp(-x) -> inf for very negative x => rcp -> 0  (correct asymptote)
    return fast_rcp(1.0f + __expf(-x));
}
__device__ __forceinline__ float tanh_f(float x) {
    // tanh(x) = 1 - 2/(exp(2x)+1); exact saturation at +/-1, no clamp needed
    float e = __expf(2.0f * x);
    return 1.0f - 2.0f * fast_rcp(e + 1.0f);
}

// A-fragment (16x32 f16, M = lane%16). ISA layout: lanes 0-15 hold K=[0..7] in
// v0-3 and K=[16..23] in v4-7; lanes 16-31 hold K=[8..15] / K=[24..31].
__device__ __forceinline__ v16h load_a_frag(const _Float16* base, int row0,
                                            int stride, int k0) {
    const int lane = threadIdx.x & 31;
    const int m  = lane & 15;
    const int hh = lane >> 4;
    const _Float16* p = base + (row0 + m) * stride + k0 + 8 * hh;
    v8h lo = *(const v8h*)(p);
    v8h hi = *(const v8h*)(p + 16);
    return __builtin_shufflevector(lo, hi, 0, 1, 2, 3, 4, 5, 6, 7,
                                   8, 9, 10, 11, 12, 13, 14, 15);
}

// B-fragment (32x16 f16, N = lane%16, weights stored N-major [gate][K]).
// ISA layout: lanes 0-15 hold K=[0..7]/K=[16..23], lanes 16-31 hold K=[8..15]/K=[24..31].
__device__ __forceinline__ v16h load_b_frag(const _Float16* base, int row0,
                                            int stride, int k0) {
    const int lane = threadIdx.x & 31;
    const int n  = lane & 15;
    const int hh = lane >> 4;
    const _Float16* p = base + (row0 + n) * stride + k0 + 8 * hh;
    v8h lo = *(const v8h*)(p);
    v8h hi = *(const v8h*)(p + 16);
    return __builtin_shufflevector(lo, hi, 0, 1, 2, 3, 4, 5, 6, 7,
                                   8, 9, 10, 11, 12, 13, 14, 15);
}

// Accumulate gates for this wave's 16 hidden columns over both M-tiles.
// acc[mt][q], q = i,f,g,o -> weight rows q*128 + wave*16 + n.
__device__ __forceinline__ void gemm_gates(const _Float16* A, int astride, int K,
                                           const _Float16* W, int wave,
                                           v8f acc[2][4]) {
    for (int k0 = 0; k0 < K; k0 += 32) {
        v16h b[4];
#pragma unroll
        for (int q = 0; q < 4; ++q)
            b[q] = load_b_frag(W, q * 128 + wave * 16, K, k0);
#pragma unroll
        for (int mt = 0; mt < 2; ++mt) {
            v16h a = load_a_frag(A, mt * 16, astride, k0);
#pragma unroll
            for (int q = 0; q < 4; ++q)
                acc[mt][q] = __builtin_amdgcn_wmma_f32_16x16x32_f16(
                    false, a, false, b[q], (short)0, acc[mt][q], false, false);
        }
    }
}

__device__ __forceinline__ void zero_acc(v8f acc[2][4]) {
#pragma unroll
    for (int mt = 0; mt < 2; ++mt)
#pragma unroll
        for (int q = 0; q < 4; ++q)
            acc[mt][q] = (v8f)(0.0f);   // first WMMA of chain uses inline-0 C
}

// ---------------------------------------------------------------------------
// Prep kernel 1: f32 -> f16 weight conversion + fused biases
// ---------------------------------------------------------------------------
__global__ void prep_kernel(const float* __restrict__ Wih0, const float* __restrict__ Whh0,
                            const float* __restrict__ Wih1, const float* __restrict__ Whh1,
                            const float* __restrict__ bih0, const float* __restrict__ bhh0,
                            const float* __restrict__ bih1, const float* __restrict__ bhh1,
                            _Float16* __restrict__ Wih0h, _Float16* __restrict__ Whh0h,
                            _Float16* __restrict__ Wih1h, _Float16* __restrict__ Whh1h,
                            float* __restrict__ bias0, float* __restrict__ bias1) {
    int i = blockIdx.x * blockDim.x + threadIdx.x;    // 0 .. 65535
    if (i < 512 * 64) Wih0h[i] = (_Float16)Wih0[i];
    if (i < 512 * 128) {
        Whh0h[i] = (_Float16)Whh0[i];
        Wih1h[i] = (_Float16)Wih1[i];
        Whh1h[i] = (_Float16)Whh1[i];
    }
    if (i < 512) {
        bias0[i] = bih0[i] + bhh0[i];
        bias1[i] = bih1[i] + bhh1[i];
    }
}

// ---------------------------------------------------------------------------
// Prep kernel 2: mu-gates, mug[b][g] = sum_k mu[b][k] * Wih0[g][k] (per batch)
// ---------------------------------------------------------------------------
__global__ void mug_kernel(const float* __restrict__ mu, const float* __restrict__ Wih0,
                           float* __restrict__ mug) {
    int i = blockIdx.x * blockDim.x + threadIdx.x;    // 0 .. 32767
    int b = i >> 9;
    int g = i & 511;
    float s = 0.0f;
#pragma unroll 8
    for (int k = 0; k < 64; ++k) s += mu[b * 64 + k] * Wih0[g * 64 + k];
    mug[i] = s;
}

// ---------------------------------------------------------------------------
// Main kernel: 32 instances / block, 8 waves, all GEMMs via WMMA f16->f32
// ---------------------------------------------------------------------------
__global__ __launch_bounds__(256) void lstm2_kernel(
    const float* __restrict__ x, const float* __restrict__ fc_w,
    const float* __restrict__ fc_b,
    const _Float16* __restrict__ Wih0h, const _Float16* __restrict__ Whh0h,
    const _Float16* __restrict__ Wih1h, const _Float16* __restrict__ Whh1h,
    const float* __restrict__ bias0, const float* __restrict__ bias1,
    const float* __restrict__ mug, float* __restrict__ out) {
    __shared__ __align__(16) _Float16 xs[32 * 72];    // step-1 inputs (f16)
    __shared__ __align__(16) _Float16 hA[32 * 136];   // layer0 h1
    __shared__ __align__(16) _Float16 hB[32 * 136];   // layer0 h2
    __shared__ __align__(16) _Float16 hC[32 * 136];   // layer1 h1'
    __shared__ __align__(16) float    fbuf[32 * 132]; // relu(h2') f32
    __shared__ float psum[32 * 8];

    const int tid  = threadIdx.x;
    const int lane = tid & 31;
    const int wave = tid >> 5;
    const int n    = lane & 15;
    const int hh   = lane >> 4;
    const long inst0 = (long)blockIdx.x * 32;
    const int  batch = (int)(inst0 >> 10);            // S = 1024 per batch
    const int  col   = wave * 16 + n;                 // this lane's hidden col

    // Stage x tile (32 x 64 f32) -> f16 LDS
    for (int i = tid; i < 32 * 64; i += 256) {
        int m = i >> 6, k = i & 63;
        xs[m * 72 + k] = (_Float16)x[(inst0 + m) * 64 + k];
    }
    __syncthreads();

    float b0v[4], b1v[4], bmv[4];
#pragma unroll
    for (int q = 0; q < 4; ++q) {
        b0v[q] = bias0[q * 128 + col];
        b1v[q] = bias1[q * 128 + col];
        // layer-0 step-2 constant: bias0 + mu@Wih0^T (per batch)
        bmv[q] = b0v[q] + mug[batch * 512 + q * 128 + col];
    }

    v8f acc[2][4];
    float c0s[2][8];   // layer-0 cell state for this wave's columns
    float c1s[2][8];   // layer-1 cell state

    // ---- Layer 0, step 1: g = x1 @ Wih0^T + bias0  (h0 = c0 = 0) ----
    zero_acc(acc);
    gemm_gates(xs, 72, 64, Wih0h, wave, acc);
#pragma unroll
    for (int mt = 0; mt < 2; ++mt)
#pragma unroll
        for (int r = 0; r < 8; ++r) {
            float ig = sigmoid_f(acc[mt][0][r] + b0v[0]);
            float gg = tanh_f(acc[mt][2][r] + b0v[2]);
            float og = sigmoid_f(acc[mt][3][r] + b0v[3]);
            float cn = ig * gg;                      // f-gate * 0
            c0s[mt][r] = cn;
            hA[(mt * 16 + r + 8 * hh) * 136 + col] = (_Float16)(og * tanh_f(cn));
        }
    __syncthreads();

    // ---- Layer 0, step 2: g = mu@Wih0^T + h1@Whh0^T + bias0 ----
    zero_acc(acc);
    gemm_gates(hA, 136, 128, Whh0h, wave, acc);
#pragma unroll
    for (int mt = 0; mt < 2; ++mt)
#pragma unroll
        for (int r = 0; r < 8; ++r) {
            float ig = sigmoid_f(acc[mt][0][r] + bmv[0]);
            float fg = sigmoid_f(acc[mt][1][r] + bmv[1]);
            float gg = tanh_f(acc[mt][2][r] + bmv[2]);
            float og = sigmoid_f(acc[mt][3][r] + bmv[3]);
            float cn = fg * c0s[mt][r] + ig * gg;
            hB[(mt * 16 + r + 8 * hh) * 136 + col] = (_Float16)(og * tanh_f(cn));
        }
    __syncthreads();

    // ---- Layer 1, step 1: g = h1 @ Wih1^T + bias1 ----
    zero_acc(acc);
    gemm_gates(hA, 136, 128, Wih1h, wave, acc);
#pragma unroll
    for (int mt = 0; mt < 2; ++mt)
#pragma unroll
        for (int r = 0; r < 8; ++r) {
            float ig = sigmoid_f(acc[mt][0][r] + b1v[0]);
            float gg = tanh_f(acc[mt][2][r] + b1v[2]);
            float og = sigmoid_f(acc[mt][3][r] + b1v[3]);
            float cn = ig * gg;
            c1s[mt][r] = cn;
            hC[(mt * 16 + r + 8 * hh) * 136 + col] = (_Float16)(og * tanh_f(cn));
        }
    __syncthreads();

    // ---- Layer 1, step 2: g = h2 @ Wih1^T + h1' @ Whh1^T + bias1 ----
    zero_acc(acc);
    gemm_gates(hB, 136, 128, Wih1h, wave, acc);
    gemm_gates(hC, 136, 128, Whh1h, wave, acc);
#pragma unroll
    for (int mt = 0; mt < 2; ++mt)
#pragma unroll
        for (int r = 0; r < 8; ++r) {
            float ig = sigmoid_f(acc[mt][0][r] + b1v[0]);
            float fg = sigmoid_f(acc[mt][1][r] + b1v[1]);
            float gg = tanh_f(acc[mt][2][r] + b1v[2]);
            float og = sigmoid_f(acc[mt][3][r] + b1v[3]);
            float cn = fg * c1s[mt][r] + ig * gg;
            float hv = og * tanh_f(cn);
            fbuf[(mt * 16 + r + 8 * hh) * 132 + col] = fmaxf(hv, 0.0f);
        }
    __syncthreads();

    // ---- FC head: out = relu(h2') . fc_w + fc_b ----
    {
        int inst = tid >> 3;     // 0..31
        int seg  = tid & 7;      // 16 columns each
        float s = 0.0f;
#pragma unroll
        for (int c = 0; c < 16; ++c) {
            int cc = seg * 16 + c;
            s += fbuf[inst * 132 + cc] * fc_w[cc];
        }
        psum[inst * 8 + seg] = s;
    }
    __syncthreads();
    if (tid < 32) {
        float s = fc_b[0];
#pragma unroll
        for (int q = 0; q < 8; ++q) s += psum[tid * 8 + q];
        out[inst0 + tid] = s;
    }
}

// ---------------------------------------------------------------------------
// Launch
// ---------------------------------------------------------------------------
extern "C" void kernel_launch(void* const* d_in, const int* in_sizes, int n_in,
                              void* d_out, int out_size, void* d_ws, size_t ws_size,
                              hipStream_t stream) {
    const float* x     = (const float*)d_in[0];
    const float* mu    = (const float*)d_in[1];
    const float* Wih0  = (const float*)d_in[2];
    const float* Whh0  = (const float*)d_in[3];
    const float* bih0  = (const float*)d_in[4];
    const float* bhh0  = (const float*)d_in[5];
    const float* Wih1  = (const float*)d_in[6];
    const float* Whh1  = (const float*)d_in[7];
    const float* bih1  = (const float*)d_in[8];
    const float* bhh1  = (const float*)d_in[9];
    const float* fc_w  = (const float*)d_in[10];
    const float* fc_b  = (const float*)d_in[11];
    float* out = (float*)d_out;

    // Workspace layout (bytes)
    char* ws = (char*)d_ws;
    _Float16* Wih0h = (_Float16*)(ws + 0);        // 512*64  f16 =  65536 B
    _Float16* Whh0h = (_Float16*)(ws + 65536);    // 512*128 f16 = 131072 B
    _Float16* Wih1h = (_Float16*)(ws + 196608);   // 131072 B
    _Float16* Whh1h = (_Float16*)(ws + 327680);   // 131072 B
    float*    bias0 = (float*)(ws + 458752);      // 512 f32
    float*    bias1 = (float*)(ws + 460800);      // 512 f32
    float*    mug   = (float*)(ws + 462848);      // 64*512 f32 = 131072 B

    prep_kernel<<<256, 256, 0, stream>>>(Wih0, Whh0, Wih1, Whh1,
                                         bih0, bhh0, bih1, bhh1,
                                         Wih0h, Whh0h, Wih1h, Whh1h,
                                         bias0, bias1);
    mug_kernel<<<128, 256, 0, stream>>>(mu, Wih0, mug);

    // 65536 instances, 32 per block
    lstm2_kernel<<<2048, 256, 0, stream>>>(x, fc_w, fc_b,
                                           Wih0h, Whh0h, Wih1h, Whh1h,
                                           bias0, bias1, mug, out);
}
